// BigramTransformer_76407468196218
// MI455X (gfx1250) — matrix operand order repaired and checked by reading, not compiled, not run on hardware
//
#include <hip/hip_runtime.h>
#include <hip/hip_fp16.h>
#include <math.h>

// Model dims (fixed by reference)
#define B_  4
#define T_  1024
#define E_  1024
#define H_  16
#define DH_ 64
#define L_  6
#define V_  32000
#define FF_ 4096
#define M_  (B_*T_)   // 4096 token rows

typedef __attribute__((ext_vector_type(16))) _Float16     v16h;
typedef __attribute__((ext_vector_type(8)))  _Float16     v8h;
typedef __attribute__((ext_vector_type(8)))  float        v8f;
typedef __attribute__((ext_vector_type(4)))  unsigned int u32x4;
typedef __attribute__((ext_vector_type(8)))  int          i32x8;
typedef __attribute__((ext_vector_type(4)))  int          i32x4;

#if defined(__has_builtin)
#if __has_builtin(__builtin_amdgcn_tensor_load_to_lds) && \
    __has_builtin(__builtin_amdgcn_s_wait_tensorcnt)
#define USE_TDM 1
#endif
#endif
#ifndef USE_TDM
#define USE_TDM 0
#endif

__device__ __forceinline__ v8f wmma_f16(v16h a, v16h b, v8f c) {
  // D = A(16x32 f16) x B(32x16 f16) + C(16x16 f32)
  return __builtin_amdgcn_wmma_f32_16x16x32_f16(false, a, false, b, (short)0, c,
                                                false, false);
}

#define NEG_BIG (-3.0e38f)

#if USE_TDM
// ---------------------------------------------------------------------------
// TDM: DMA a [rows x 32]-half tile (row stride = strideHalves in global) into
// LDS with 8 halves of padding per row (pad 4 DWORDs every 16 DWORDs) so the
// LDS image is rows x 40 halves, matching the fragment-read layout.
// ---------------------------------------------------------------------------
__device__ __forceinline__ void tdm_load_tile_f16(const _Float16* gsrc,
                                                  unsigned lds_byte_off,
                                                  int rows,
                                                  long long strideHalves) {
  const unsigned long long ga = (unsigned long long)(size_t)gsrc;
  u32x4 g0;
  g0[0] = 1u;                                          // count=1 user D#
  g0[1] = lds_byte_off;                                // lds_addr (bytes)
  g0[2] = (unsigned)(ga & 0xFFFFFFFFull);              // global_addr[31:0]
  g0[3] = (unsigned)((ga >> 32) & 0x1FFFFFFull)        // global_addr[56:32]
        | (2u << 30);                                  // type = 2 (image)
  i32x8 g1;
  g1[0] = (1 << 16)                                    // data_size = 2B
        | (1 << 20)                                    // pad_enable
        | (3 << 22)                                    // pad_interval = 16 DW
        | (3 << 25);                                   // pad_amount  = 4 DW
  g1[1] = (32 & 0xFFFF) << 16;                         // tensor_dim0 lo (=32)
  g1[2] = (int)(((unsigned)rows & 0xFFFFu) << 16);     // dim0 hi=0 | dim1 lo
  g1[3] = (int)(32u << 16);                            // dim1 hi=0 | tile_dim0
  g1[4] = rows & 0xFFFF;                               // tile_dim1 | tile_dim2=0
  g1[5] = (int)(unsigned)(strideHalves & 0xFFFFFFFFll);// dim0_stride lo
  g1[6] = (int)((strideHalves >> 32) & 0xFFFFll);      // dim0_stride hi
  g1[7] = 0;
  i32x4 z4; z4[0] = 0; z4[1] = 0; z4[2] = 0; z4[3] = 0;
#if defined(__clang_major__) && (__clang_major__ >= 23)
  i32x8 z8;
#pragma unroll
  for (int i = 0; i < 8; ++i) z8[i] = 0;
  __builtin_amdgcn_tensor_load_to_lds(g0, g1, z4, z4, z8, 0);
#else
  __builtin_amdgcn_tensor_load_to_lds(g0, g1, z4, z4, 0);
#endif
}
#endif  // USE_TDM

// ---------------------------------------------------------------------------
// fp32 -> fp16 bulk convert (weights pre-pass). n must be a multiple of 2048.
// ---------------------------------------------------------------------------
__global__ __launch_bounds__(256)
void cvt_f16_kernel(const float* __restrict__ src, _Float16* __restrict__ dst) {
  const long long i = ((long long)blockIdx.x * 256 + threadIdx.x) * 8;
  const float4 a = *(const float4*)(src + i);
  const float4 b = *(const float4*)(src + i + 4);
  v8h o;
  o[0] = (_Float16)a.x; o[1] = (_Float16)a.y; o[2] = (_Float16)a.z; o[3] = (_Float16)a.w;
  o[4] = (_Float16)b.x; o[5] = (_Float16)b.y; o[6] = (_Float16)b.z; o[7] = (_Float16)b.w;
  *(v8h*)(dst + i) = o;
}

// ---------------------------------------------------------------------------
// Embedding: h[row, :] = tok_emb[x[row], :] + pos_emb[row % T, :]
// ---------------------------------------------------------------------------
__global__ __launch_bounds__(256)
void embed_kernel(const int* __restrict__ x, const float* __restrict__ tok,
                  const float* __restrict__ pos, float* __restrict__ h) {
  const int row = blockIdx.x;
  const int t   = row & (T_ - 1);
  const int id  = x[row];
  const float4* tv = (const float4*)(tok + (size_t)id * E_);
  const float4* pv = (const float4*)(pos + (size_t)t  * E_);
  float4* hv = (float4*)(h + (size_t)row * E_);
  for (int i = threadIdx.x; i < E_ / 4; i += 256) {
    float4 a = tv[i], b = pv[i];
    float4 o; o.x = a.x + b.x; o.y = a.y + b.y; o.z = a.z + b.z; o.w = a.w + b.w;
    hv[i] = o;
  }
}

// ---------------------------------------------------------------------------
// LayerNorm over E=1024 (population var, eps=1e-5). f32 in, f16 out (output
// only ever feeds WMMA GEMMs as the A operand).
// ---------------------------------------------------------------------------
__global__ __launch_bounds__(256)
void ln_kernel(const float* __restrict__ in, const float* __restrict__ g,
               const float* __restrict__ bb, _Float16* __restrict__ out) {
  const int row = blockIdx.x;
  const int tid = threadIdx.x;
  const float* xr = in + (size_t)row * E_;
  float vals[4];
  float s = 0.f, s2 = 0.f;
#pragma unroll
  for (int j = 0; j < 4; ++j) {
    float v = xr[tid + j * 256];
    vals[j] = v; s += v; s2 += v * v;
  }
#pragma unroll
  for (int off = 16; off > 0; off >>= 1) {
    s  += __shfl_down(s,  off, 32);
    s2 += __shfl_down(s2, off, 32);
  }
  __shared__ float sh1[8], sh2[8];
  const int wid = tid >> 5, lane = tid & 31;
  if (lane == 0) { sh1[wid] = s; sh2[wid] = s2; }
  __syncthreads();
  if (tid == 0) {
    float a = 0.f, b = 0.f;
#pragma unroll
    for (int i = 0; i < 8; ++i) { a += sh1[i]; b += sh2[i]; }
    sh1[0] = a; sh2[0] = b;
  }
  __syncthreads();
  const float mean = sh1[0] * (1.f / E_);
  const float var  = sh2[0] * (1.f / E_) - mean * mean;
  const float rstd = rsqrtf(var + 1e-5f);
#pragma unroll
  for (int j = 0; j < 4; ++j) {
    const int c = tid + j * 256;
    out[(size_t)row * E_ + c] = (_Float16)((vals[j] - mean) * rstd * g[c] + bb[c]);
  }
}

// ---------------------------------------------------------------------------
// WMMA GEMM: C[M,N] = act( A[M,K](f16) * W[N,K](f16)^T + bias ) (+ res)
// Block tile 128x64, 8 waves -> 32x32 per wave. TDM double-buffered staging.
// ---------------------------------------------------------------------------
__global__ __launch_bounds__(256)
void gemm_wmma(const _Float16* __restrict__ A, const _Float16* __restrict__ W,
               const float* __restrict__ bias, const float* __restrict__ res,
               float* __restrict__ C32, _Float16* __restrict__ C16,
               int Ndim, int Kdim, int relu) {
#if USE_TDM
  __shared__ __align__(16) _Float16 As[2][128][40];
  __shared__ __align__(16) _Float16 Ws[2][64][40];
#else
  __shared__ __align__(16) _Float16 As[1][128][40];
  __shared__ __align__(16) _Float16 Ws[1][64][40];
#endif
  const int tid  = threadIdx.x;
  const int lane = tid & 31, wid = tid >> 5;
  const int half = (lane >> 4) & 1, lr = lane & 15;
  const int waveM = wid & 3, waveN = wid >> 2;   // 4 x 2 wave grid
  const int m0 = blockIdx.y * 128, n0 = blockIdx.x * 64;

  v8f acc[2][2];
#pragma unroll
  for (int i = 0; i < 2; ++i)
#pragma unroll
    for (int j = 0; j < 2; ++j)
#pragma unroll
      for (int r = 0; r < 8; ++r) acc[i][j][r] = 0.f;

#if USE_TDM
  // Prologue: DMA first A/W tiles into buffer 0.
  if (wid == 0) {
    tdm_load_tile_f16(A + (size_t)m0 * Kdim, (unsigned)(size_t)&As[0][0][0],
                      128, Kdim);
    tdm_load_tile_f16(W + (size_t)n0 * Kdim, (unsigned)(size_t)&Ws[0][0][0],
                      64, Kdim);
  }
  int cur = 0;
#endif

  for (int k0 = 0; k0 < Kdim; k0 += 32) {
#if USE_TDM
    const int nxt = cur ^ 1;
    if (wid == 0) {
      if (k0 + 32 < Kdim) {
        // Overlap: DMA tile k0+32 while tile k0 is consumed below.
        tdm_load_tile_f16(A + (size_t)m0 * Kdim + (k0 + 32),
                          (unsigned)(size_t)&As[nxt][0][0], 128, Kdim);
        tdm_load_tile_f16(W + (size_t)n0 * Kdim + (k0 + 32),
                          (unsigned)(size_t)&Ws[nxt][0][0], 64, Kdim);
        __builtin_amdgcn_s_wait_tensorcnt((short)2);  // tile k0 complete
      } else {
        __builtin_amdgcn_s_wait_tensorcnt((short)0);
      }
    }
    __syncthreads();                      // LDS buffer `cur` valid for all waves
#define AS As[cur]
#define WS Ws[cur]
#else
    __syncthreads();
    // Manual staging fallback: issue both loads before the LDS stores.
    {
      const int r = tid >> 1, c = (tid & 1) * 16;           // A: 128x2 groups
      const v8h a0 = *(const v8h*)(A + (size_t)(m0 + r) * Kdim + k0 + c);
      const v8h a1 = *(const v8h*)(A + (size_t)(m0 + r) * Kdim + k0 + c + 8);
      v8h w0, w1; const int rw = tid >> 1, cw = (tid & 1) * 16;
      if (tid < 128) {                                       // W: 64x2 groups
        w0 = *(const v8h*)(W + (size_t)(n0 + rw) * Kdim + k0 + cw);
        w1 = *(const v8h*)(W + (size_t)(n0 + rw) * Kdim + k0 + cw + 8);
      }
      *(v8h*)&As[0][r][c] = a0; *(v8h*)&As[0][r][c + 8] = a1;
      if (tid < 128) { *(v8h*)&Ws[0][rw][cw] = w0; *(v8h*)&Ws[0][rw][cw + 8] = w1; }
    }
    __syncthreads();
#define AS As[0]
#define WS Ws[0]
#endif

    // A frag (16x32): lane row = lr; k(e) = (e>=8?16:0) + half*8 + (e&7)
    v16h af[2];
#pragma unroll
    for (int ms = 0; ms < 2; ++ms) {
      const int row = waveM * 32 + ms * 16 + lr;
      const v8h lo = *(const v8h*)&AS[row][half * 8];
      const v8h hi = *(const v8h*)&AS[row][16 + half * 8];
#pragma unroll
      for (int e = 0; e < 8; ++e) { af[ms][e] = lo[e]; af[ms][e + 8] = hi[e]; }
    }
    // B frag (32x16): lane col = lr; k(e) = (e>=8?16:0) + half*8 + (e&7)
    v16h bf[2];
#pragma unroll
    for (int ns = 0; ns < 2; ++ns) {
      const int row = waveN * 32 + ns * 16 + lr;
      const v8h lo = *(const v8h*)&WS[row][half * 8];
      const v8h hi = *(const v8h*)&WS[row][16 + half * 8];
#pragma unroll
      for (int e = 0; e < 8; ++e) { bf[ns][e] = lo[e]; bf[ns][e + 8] = hi[e]; }
    }
#pragma unroll
    for (int ms = 0; ms < 2; ++ms)
#pragma unroll
      for (int ns = 0; ns < 2; ++ns)
        acc[ms][ns] = wmma_f16(af[ms], bf[ns], acc[ms][ns]);

#if USE_TDM
    __syncthreads();   // all waves done with `cur` before its next DMA reuse
    cur = nxt;
#endif
#undef AS
#undef WS
  }

  // Fused epilogue: bias + residual + ReLU. C frag: M = r + half*8, N = lr.
#pragma unroll
  for (int ms = 0; ms < 2; ++ms)
#pragma unroll
    for (int ns = 0; ns < 2; ++ns) {
      const int gn = n0 + waveN * 32 + ns * 16 + lr;
      const float bv = bias ? bias[gn] : 0.f;
#pragma unroll
      for (int r = 0; r < 8; ++r) {
        const int gm = m0 + waveM * 32 + ms * 16 + r + half * 8;
        float v = acc[ms][ns][r] + bv;
        if (res)  v += res[(size_t)gm * Ndim + gn];
        if (relu) v = fmaxf(v, 0.f);
        if (C32)  C32[(size_t)gm * Ndim + gn] = v;
        if (C16)  C16[(size_t)gm * Ndim + gn] = (_Float16)v;
      }
    }
}

// ---------------------------------------------------------------------------
// Flash attention: 1 wave per (b, h, 16-query tile); online softmax.
// Q/K/V f16, laid out [B,T,H*DH] (column = h*64 + d); O written as f16.
// ---------------------------------------------------------------------------
__global__ __launch_bounds__(32)
void attn_kernel(const _Float16* __restrict__ Q, const _Float16* __restrict__ Kt,
                 const _Float16* __restrict__ Vt, _Float16* __restrict__ O) {
  const int lane = threadIdx.x;
  const int half = lane >> 4, lr = lane & 15;
  const int q0 = blockIdx.x * 16;
  const int h  = blockIdx.y;
  const int b  = blockIdx.z;
  const float scale = 0.125f;  // DH^-0.5

  __shared__ __align__(16) _Float16 Pld[16][40];

  // Q fragments (A layout), pre-scaled by 1/sqrt(DH)
  v16h qf[2];
#pragma unroll
  for (int kk = 0; kk < 2; ++kk) {
    const _Float16* qrow =
        Q + ((size_t)b * T_ + (q0 + lr)) * E_ + h * DH_ + kk * 32 + half * 8;
    const v8h lo = *(const v8h*)qrow;            // k = kk*32 + half*8 + e
    const v8h hi = *(const v8h*)(qrow + 16);     // k = ... + 16
#pragma unroll
    for (int e = 0; e < 8; ++e) {
      qf[kk][e]     = (_Float16)((float)lo[e] * scale);
      qf[kk][e + 8] = (_Float16)((float)hi[e] * scale);
    }
  }

  float mstate[8], lstate[8];
#pragma unroll
  for (int r = 0; r < 8; ++r) { mstate[r] = NEG_BIG; lstate[r] = 0.f; }
  v8f oacc[4];
#pragma unroll
  for (int j = 0; j < 4; ++j)
#pragma unroll
    for (int r = 0; r < 8; ++r) oacc[j][r] = 0.f;

  for (int s0 = 0; s0 < q0 + 16; s0 += 32) {   // 32-key chunks, causal bound
    if (s0 + 32 < q0 + 16) {                   // prefetch next key chunk
      __builtin_prefetch(Kt + ((size_t)b * T_ + (s0 + 32 + lane)) * E_ + h * DH_, 0, 0);
      __builtin_prefetch(Vt + ((size_t)b * T_ + (s0 + 32 + lane)) * E_ + h * DH_, 0, 0);
    }
    // ---- S = Q K^T (two 16x16 tiles over this chunk) ----
    v8f sacc[2];
#pragma unroll
    for (int ns = 0; ns < 2; ++ns)
#pragma unroll
      for (int r = 0; r < 8; ++r) sacc[ns][r] = 0.f;
#pragma unroll
    for (int ns = 0; ns < 2; ++ns) {
      int key = s0 + ns * 16 + lr; if (key > T_ - 1) key = T_ - 1;  // masked below
#pragma unroll
      for (int kk = 0; kk < 2; ++kk) {
        const _Float16* krow =
            Kt + ((size_t)b * T_ + key) * E_ + h * DH_ + kk * 32 + half * 8;
        const v8h lo = *(const v8h*)krow;        // k(e) = (e>=8?16:0) + half*8 + (e&7)
        const v8h hi = *(const v8h*)(krow + 16);
        v16h bfr;
#pragma unroll
        for (int e = 0; e < 8; ++e) { bfr[e] = lo[e]; bfr[e + 8] = hi[e]; }
        sacc[ns] = wmma_f16(qf[kk], bfr, sacc[ns]);
      }
    }
    // ---- causal mask + online softmax (rows r+half*8 live in this half) ----
    float sm[2][8];
#pragma unroll
    for (int ns = 0; ns < 2; ++ns) {
      const int kn = s0 + ns * 16 + lr;
#pragma unroll
      for (int r = 0; r < 8; ++r) {
        const int qm = q0 + r + half * 8;
        sm[ns][r] = (kn <= qm) ? sacc[ns][r] : NEG_BIG;
      }
    }
#pragma unroll
    for (int r = 0; r < 8; ++r) {
      float rm = fmaxf(sm[0][r], sm[1][r]);
#pragma unroll
      for (int off = 1; off < 16; off <<= 1) rm = fmaxf(rm, __shfl_xor(rm, off, 32));
      const float newm = fmaxf(mstate[r], rm);
      const float corr = __expf(mstate[r] - newm);
      const float p0 = __expf(sm[0][r] - newm);
      const float p1 = __expf(sm[1][r] - newm);
      float ps = p0 + p1;
#pragma unroll
      for (int off = 1; off < 16; off <<= 1) ps += __shfl_xor(ps, off, 32);
      lstate[r] = lstate[r] * corr + ps;
      mstate[r] = newm;
#pragma unroll
      for (int j = 0; j < 4; ++j) oacc[j][r] *= corr;
      Pld[r + half * 8][lr]      = (_Float16)p0;   // S tile ns=0
      Pld[r + half * 8][16 + lr] = (_Float16)p1;   // S tile ns=1
    }
    __syncthreads();
    // ---- re-layout P: C-frag -> A-frag via LDS ----
    v16h pf;
    {
      const v8h lo = *(const v8h*)&Pld[lr][half * 8];
      const v8h hi = *(const v8h*)&Pld[lr][16 + half * 8];
#pragma unroll
      for (int e = 0; e < 8; ++e) { pf[e] = lo[e]; pf[e + 8] = hi[e]; }
    }
    __syncthreads();
    // ---- O += P V (N = DH in 4 column tiles) ----
#pragma unroll
    for (int j = 0; j < 4; ++j) {
      v16h vfr;
#pragma unroll
      for (int e = 0; e < 16; ++e) {
        int key = s0 + (e >= 8 ? 16 : 0) + half * 8 + (e & 7); if (key > T_ - 1) key = T_ - 1;  // p==0 there
        vfr[e] = Vt[((size_t)b * T_ + key) * E_ + h * DH_ + j * 16 + lr];
      }
      oacc[j] = wmma_f16(pf, vfr, oacc[j]);
    }
  }
  // ---- normalize and write concat-head output (f16) ----
#pragma unroll
  for (int j = 0; j < 4; ++j)
#pragma unroll
    for (int r = 0; r < 8; ++r) {
      const int gm = q0 + r + half * 8;
      O[((size_t)b * T_ + gm) * E_ + h * DH_ + j * 16 + lr] =
          (_Float16)(oacc[j][r] / lstate[r]);
    }
}

// ---------------------------------------------------------------------------
// Per-row cross-entropy: loss[row] = logsumexp(logits[row]) - logits[row,tgt]
// ---------------------------------------------------------------------------
__global__ __launch_bounds__(256)
void loss_row_kernel(const float* __restrict__ logits, const int* __restrict__ tgt,
                     float* __restrict__ rowloss) {
  const int row = blockIdx.x;
  const int tid = threadIdx.x;
  const float* lg = logits + (size_t)row * V_;
  __shared__ float shm[8], shs[8];
  float mx = NEG_BIG;
  for (int i = tid; i < V_; i += 256) mx = fmaxf(mx, lg[i]);
#pragma unroll
  for (int off = 16; off > 0; off >>= 1) mx = fmaxf(mx, __shfl_xor(mx, off, 32));
  const int wid = tid >> 5, lane = tid & 31;
  if (lane == 0) shm[wid] = mx;
  __syncthreads();
  if (tid == 0) {
    float m = shm[0];
#pragma unroll
    for (int i = 1; i < 8; ++i) m = fmaxf(m, shm[i]);
    shm[0] = m;
  }
  __syncthreads();
  mx = shm[0];
  float s = 0.f;
  for (int i = tid; i < V_; i += 256) s += __expf(lg[i] - mx);
#pragma unroll
  for (int off = 16; off > 0; off >>= 1) s += __shfl_xor(s, off, 32);
  if (lane == 0) shs[wid] = s;
  __syncthreads();
  if (tid == 0) {
    float a = 0.f;
#pragma unroll
    for (int i = 0; i < 8; ++i) a += shs[i];
    rowloss[row] = (mx + logf(a)) - lg[tgt[row]];
  }
}

__global__ __launch_bounds__(256)
void loss_reduce_kernel(const float* __restrict__ rowloss, float* __restrict__ out) {
  const int tid = threadIdx.x;
  float s = 0.f;
  for (int i = tid; i < M_; i += 256) s += rowloss[i];
#pragma unroll
  for (int off = 16; off > 0; off >>= 1) s += __shfl_xor(s, off, 32);
  __shared__ float sh[8];
  const int wid = tid >> 5, lane = tid & 31;
  if (lane == 0) sh[wid] = s;
  __syncthreads();
  if (tid == 0) {
    float a = 0.f;
#pragma unroll
    for (int i = 0; i < 8; ++i) a += sh[i];
    out[0] = a * (1.f / M_);
  }
}

// ---------------------------------------------------------------------------
// Host-side orchestration (stream-only; graph-capture safe)
// ---------------------------------------------------------------------------
extern "C" void kernel_launch(void* const* d_in, const int* in_sizes, int n_in,
                              void* d_out, int out_size, void* d_ws, size_t ws_size,
                              hipStream_t stream) {
  (void)in_sizes; (void)n_in; (void)out_size; (void)ws_size;
  const int*   x    = (const int*)  d_in[0];
  const int*   tgt  = (const int*)  d_in[1];
  const float* tok  = (const float*)d_in[2];
  const float* pos  = (const float*)d_in[3];
  const float* Wq   = (const float*)d_in[4];
  const float* Wk   = (const float*)d_in[5];
  const float* Wv   = (const float*)d_in[6];
  const float* Wo   = (const float*)d_in[7];
  const float* bo   = (const float*)d_in[8];
  const float* ln1g = (const float*)d_in[9];
  const float* ln1b = (const float*)d_in[10];
  const float* ln2g = (const float*)d_in[11];
  const float* ln2b = (const float*)d_in[12];
  const float* W1   = (const float*)d_in[13];
  const float* b1   = (const float*)d_in[14];
  const float* W2   = (const float*)d_in[15];
  const float* b2   = (const float*)d_in[16];
  const float* lnfg = (const float*)d_in[17];
  const float* lnfb = (const float*)d_in[18];
  const float* Wh   = (const float*)d_in[19];
  const float* bh   = (const float*)d_in[20];

  // ---- workspace carve-out ----
  char* p = (char*)d_ws;
  float*     h   = (float*)p;     p += (size_t)M_ * E_ * 4;       // 16 MB
  float*     rl  = (float*)p;     p += (size_t)M_ * 4;            // 16 KB
  _Float16*  a16 = (_Float16*)p;  p += (size_t)M_ * E_ * 2;       // 8 MB
  _Float16*  q16 = (_Float16*)p;  p += (size_t)M_ * E_ * 2;
  _Float16*  k16 = (_Float16*)p;  p += (size_t)M_ * E_ * 2;
  _Float16*  v16 = (_Float16*)p;  p += (size_t)M_ * E_ * 2;
  _Float16*  o16 = (_Float16*)p;  p += (size_t)M_ * E_ * 2;
  _Float16*  f1h = (_Float16*)p;  p += (size_t)M_ * FF_ * 2;      // 32 MB
  _Float16*  wqh = (_Float16*)p;  p += (size_t)L_ * E_ * E_ * 2;  // 12.6 MB each
  _Float16*  wkh = (_Float16*)p;  p += (size_t)L_ * E_ * E_ * 2;
  _Float16*  wvh = (_Float16*)p;  p += (size_t)L_ * E_ * E_ * 2;
  _Float16*  woh = (_Float16*)p;  p += (size_t)L_ * E_ * E_ * 2;
  _Float16*  w1h = (_Float16*)p;  p += (size_t)L_ * FF_ * E_ * 2; // 50.3 MB
  _Float16*  w2h = (_Float16*)p;  p += (size_t)L_ * E_ * FF_ * 2;
  _Float16*  whh = (_Float16*)p;  p += (size_t)V_ * E_ * 2;       // 65.5 MB
  float* logits = (float*)d_out;
  float* loss   = logits + (size_t)M_ * V_;

  // ---- weight f32->f16 pre-pass (each size divisible by 2048) ----
  const long long nEE = (long long)L_ * E_ * E_;
  const long long nFE = (long long)L_ * FF_ * E_;
  const long long nVE = (long long)V_ * E_;
  cvt_f16_kernel<<<nEE >> 11, 256, 0, stream>>>(Wq, wqh);
  cvt_f16_kernel<<<nEE >> 11, 256, 0, stream>>>(Wk, wkh);
  cvt_f16_kernel<<<nEE >> 11, 256, 0, stream>>>(Wv, wvh);
  cvt_f16_kernel<<<nEE >> 11, 256, 0, stream>>>(Wo, woh);
  cvt_f16_kernel<<<nFE >> 11, 256, 0, stream>>>(W1, w1h);
  cvt_f16_kernel<<<nFE >> 11, 256, 0, stream>>>(W2, w2h);
  cvt_f16_kernel<<<nVE >> 11, 256, 0, stream>>>(Wh, whh);

  embed_kernel<<<M_, 256, 0, stream>>>(x, tok, pos, h);

  dim3 gEE(E_ / 64, M_ / 128);   // N=1024
  dim3 gFF(FF_ / 64, M_ / 128);  // N=4096
  dim3 gV (V_ / 64, M_ / 128);   // N=32000
  dim3 gAtt(T_ / 16, H_, B_);

  for (int l = 0; l < L_; ++l) {
    ln_kernel<<<M_, 256, 0, stream>>>(h, ln1g + l * E_, ln1b + l * E_, a16);
    gemm_wmma<<<gEE, 256, 0, stream>>>(a16, wqh + (size_t)l * E_ * E_, nullptr,
                                       nullptr, nullptr, q16, E_, E_, 0);
    gemm_wmma<<<gEE, 256, 0, stream>>>(a16, wkh + (size_t)l * E_ * E_, nullptr,
                                       nullptr, nullptr, k16, E_, E_, 0);
    gemm_wmma<<<gEE, 256, 0, stream>>>(a16, wvh + (size_t)l * E_ * E_, nullptr,
                                       nullptr, nullptr, v16, E_, E_, 0);
    attn_kernel<<<gAtt, 32, 0, stream>>>(q16, k16, v16, o16);
    gemm_wmma<<<gEE, 256, 0, stream>>>(o16, woh + (size_t)l * E_ * E_, bo + l * E_,
                                       h, h, nullptr, E_, E_, 0);
    ln_kernel<<<M_, 256, 0, stream>>>(h, ln2g + l * E_, ln2b + l * E_, a16);
    gemm_wmma<<<gFF, 256, 0, stream>>>(a16, w1h + (size_t)l * FF_ * E_, b1 + l * FF_,
                                       nullptr, nullptr, f1h, FF_, E_, 1);
    gemm_wmma<<<gEE, 256, 0, stream>>>(f1h, w2h + (size_t)l * E_ * FF_, b2 + l * E_,
                                       h, h, nullptr, E_, FF_, 0);
  }
  ln_kernel<<<M_, 256, 0, stream>>>(h, lnfg, lnfb, a16);
  gemm_wmma<<<gV, 256, 0, stream>>>(a16, whh, bh, nullptr, logits, nullptr,
                                    V_, E_, 0);
  loss_row_kernel<<<M_, 256, 0, stream>>>(logits, tgt, rl);
  loss_reduce_kernel<<<1, 256, 0, stream>>>(rl, loss);
}
